// Attention_53850299957414
// MI455X (gfx1250) — matrix operand order repaired and checked by reading, not compiled, not run on hardware
//
#include <hip/hip_runtime.h>
#include <hip/hip_bf16.h>

typedef _Float16 f16;
typedef __attribute__((ext_vector_type(16))) _Float16 v16h;
typedef __attribute__((ext_vector_type(8)))  _Float16 v8h;
typedef __attribute__((ext_vector_type(8)))  float    v8f;

#define DIMN   2048
#define HEADS  16
#define HD     128
#define SEQ    2048
#define BATCH  2
#define MTOT   (BATCH * SEQ)   // 4096 flattened (b,s) rows

// ---------------------------------------------------------------------------
// WMMA fragment loaders (gfx1250 wave32 layouts, CDNA5 ISA §7.12.2)
// A (16x32 f16): lane = {khalf, M}; elements 0..7 = K kh..kh+7, 8..15 = K 16+kh..
// B (32x16 f16): lane = {khalf, N}; elements 0..7 = K kh..kh+7, 8..15 = K 16+kh..
// C/D (16x16 f32): VGPR j holds M = j + 8*(lane>>4), N = lane&15.
// ---------------------------------------------------------------------------
static __device__ __forceinline__ v16h load_frag_a(const f16* base, int ld) {
    int lane = threadIdx.x & 31;
    int m    = lane & 15;
    int kh   = (lane >> 4) << 3;
    const f16* p = base + (size_t)m * ld + kh;
    v8h lo = *(const v8h*)p;
    v8h hi = *(const v8h*)(p + 16);
    v16h a;
#pragma unroll
    for (int i = 0; i < 8; ++i) { a[i] = lo[i]; a[i + 8] = hi[i]; }
    return a;
}

static __device__ __forceinline__ v16h load_frag_b(const f16* base, int ld) {
    int lane = threadIdx.x & 31;
    int n    = lane & 15;
    int kh   = (lane >> 4) << 3;
    const f16* p = base + (size_t)kh * ld + n;
    v16h b;
#pragma unroll
    for (int i = 0; i < 8; ++i) { b[i] = p[(size_t)i * ld]; b[i + 8] = p[(size_t)(16 + i) * ld]; }
    return b;
}

static __device__ __forceinline__ v8f wmma_f16(v16h a, v16h b, v8f c) {
    return __builtin_amdgcn_wmma_f32_16x16x32_f16(false, a, false, b,
                                                  (short)0, c, false, false);
}

// ---------------------------------------------------------------------------
// Phase 0a: fp32 x -> f16 (row-major, K contiguous: perfect A-operand layout)
// ---------------------------------------------------------------------------
__global__ __launch_bounds__(256) void cvt_x_kernel(const float* __restrict__ x,
                                                    f16* __restrict__ xh) {
    size_t i = (size_t)blockIdx.x * 256 + threadIdx.x;
    xh[i] = (f16)x[i];
}

// ---------------------------------------------------------------------------
// Phase 0b: weight transpose wT[k][n] = w[n][k], fp32 -> f16 (LDS-tiled)
// ---------------------------------------------------------------------------
__global__ __launch_bounds__(256) void cvt_w_kernel(const float* __restrict__ wq,
                                                    const float* __restrict__ wk,
                                                    const float* __restrict__ wv,
                                                    const float* __restrict__ wo,
                                                    f16* __restrict__ wqT,
                                                    f16* __restrict__ wkT,
                                                    f16* __restrict__ wvT,
                                                    f16* __restrict__ woT) {
    __shared__ float t[32][33];
    int z = blockIdx.z;
    const float* W = (z == 0) ? wq : (z == 1) ? wk : (z == 2) ? wv : wo;
    f16* D = (z == 0) ? wqT : (z == 1) ? wkT : (z == 2) ? wvT : woT;
    int k0 = blockIdx.x * 32, n0 = blockIdx.y * 32;
    int tx = threadIdx.x, ty = threadIdx.y;   // block (32,8)
#pragma unroll
    for (int i = 0; i < 4; ++i)
        t[ty + i * 8][tx] = W[(size_t)(n0 + ty + i * 8) * DIMN + k0 + tx];
    __syncthreads();
#pragma unroll
    for (int i = 0; i < 4; ++i)
        D[(size_t)(k0 + ty + i * 8) * DIMN + n0 + tx] = (f16)t[tx][ty + i * 8];
}

// ---------------------------------------------------------------------------
// Phase 1: fused QKV projection + RoPE.
//   q_h [b][h][s][d]  (f16, A-operand-ready for scores)
//   kT_h[b][h][d][s]  (f16, B-operand-ready for scores: K pre-transposed)
//   v_h [b][h][s][d]  (f16, B-operand-ready for P*V)
// grid = (M/64, N/256, 3); 8 waves -> 2x4, each wave a 32x64 tile
// (8 WMMAs per 4 B128 A loads + 64 16-bit B loads per 32-deep k-step).
// ---------------------------------------------------------------------------
__global__ __launch_bounds__(256) void qkv_rope_kernel(const f16* __restrict__ xh,
                                                       const f16* __restrict__ wqT,
                                                       const f16* __restrict__ wkT,
                                                       const f16* __restrict__ wvT,
                                                       f16* __restrict__ q_h,
                                                       f16* __restrict__ kT_h,
                                                       f16* __restrict__ v_h) {
    int mode = blockIdx.z;
    const f16* BT = (mode == 0) ? wqT : (mode == 1) ? wkT : wvT;
    int lane = threadIdx.x & 31;
    int wave = threadIdx.x >> 5;
    int m0 = blockIdx.x * 64 + (wave & 1) * 32;
    int n0 = blockIdx.y * 256 + (wave >> 1) * 64;

    v8f acc[2][4] = {};
#pragma unroll 2
    for (int k0 = 0; k0 < DIMN; k0 += 32) {
        // speculative prefetch of next k-tile of the streamed A operand
        __builtin_prefetch(xh + (size_t)m0 * DIMN + k0 + 64, 0, 0);
        v16h a0 = load_frag_a(xh + (size_t)m0 * DIMN + k0, DIMN);
        v16h a1 = load_frag_a(xh + (size_t)(m0 + 16) * DIMN + k0, DIMN);
#pragma unroll
        for (int tj = 0; tj < 4; ++tj) {
            v16h bf = load_frag_b(BT + (size_t)k0 * DIMN + n0 + tj * 16, DIMN);
            acc[0][tj] = wmma_f16(a0, bf, acc[0][tj]);
            acc[1][tj] = wmma_f16(a1, bf, acc[1][tj]);
        }
    }

    int half = lane >> 4;
#pragma unroll
    for (int ti = 0; ti < 2; ++ti)
#pragma unroll
        for (int tj = 0; tj < 4; ++tj) {
            int col = n0 + tj * 16 + (lane & 15);
            int h = col >> 7, d = col & 127;
            // RoPE inv-freq: base^{-(2i)/128}, i = d/2
            float invf = __expf(-0.07195578415606394f * (float)(d & 126));
#pragma unroll
            for (int j = 0; j < 8; ++j) {
                int row = m0 + ti * 16 + j + 8 * half;
                int bb = row >> 11, s = row & (SEQ - 1);
                float v = acc[ti][tj][j];
                size_t hb = (size_t)(bb * HEADS + h);
                if (mode == 2) {
                    v_h[(hb * SEQ + s) * HD + d] = (f16)v;
                } else {
                    float ang = (float)s * invf, sn, c;
                    __sincosf(ang, &sn, &c);
                    float other = __shfl_xor(v, 1, 32);   // pair lane (d^1)
                    float r = ((d & 1) == 0) ? (v * c - other * sn)
                                             : (v * c + other * sn);
                    if (mode == 0) q_h[(hb * SEQ + s) * HD + d] = (f16)r;
                    else           kT_h[(hb * HD + d) * SEQ + s] = (f16)r;
                }
            }
        }
}

// ---------------------------------------------------------------------------
// Phase 2: causal flash attention. One wave = 16 q-rows, online softmax,
// k-tiles of 32 (8 score WMMAs + 8 PV WMMAs per tile). P goes through a
// per-wave LDS buffer to convert C-fragment layout -> A-fragment layout.
// grid = (SEQ/128, HEADS, BATCH), 8 waves/WG.
// ---------------------------------------------------------------------------
__global__ __launch_bounds__(256) void attn_kernel(const f16* __restrict__ q_h,
                                                   const f16* __restrict__ kT_h,
                                                   const f16* __restrict__ v_h,
                                                   f16* __restrict__ a_h) {
    __shared__ __align__(16) _Float16 pbuf[8 * 16 * 32];
    int lane = threadIdx.x & 31;
    int wave = threadIdx.x >> 5;
    int half = lane >> 4;
    int q0 = blockIdx.x * 128 + wave * 16;
    int h = blockIdx.y, b = blockIdx.z;
    size_t hb = (size_t)(b * HEADS + h);
    const f16* q  = q_h  + hb * SEQ * HD;
    const f16* kT = kT_h + hb * HD * SEQ;
    const f16* vv = v_h  + hb * SEQ * HD;
    f16* pw = &pbuf[wave * 512];

    v16h qf[4];
#pragma unroll
    for (int dt = 0; dt < 4; ++dt)
        qf[dt] = load_frag_a(q + (size_t)q0 * HD + dt * 32, HD);

    v8f O[8] = {};
    float mi[8], li[8], alph[8];
#pragma unroll
    for (int j = 0; j < 8; ++j) { mi[j] = -__builtin_inff(); li[j] = 0.0f; }

    const float sc = 0.08838834764831845f;   // 1/sqrt(128)
    int jend = q0 + 16;
    for (int j0 = 0; j0 < jend; j0 += 32) {
        v8f s0 = {}, s1 = {};
#pragma unroll
        for (int dt = 0; dt < 4; ++dt) {
            v16h b0 = load_frag_b(kT + (size_t)(dt * 32) * SEQ + j0, SEQ);
            v16h b1 = load_frag_b(kT + (size_t)(dt * 32) * SEQ + j0 + 16, SEQ);
            s0 = wmma_f16(qf[dt], b0, s0);
            s1 = wmma_f16(qf[dt], b1, s1);
        }
        bool needMask = (j0 + 31 > q0);
        int colA = j0 + (lane & 15), colB = colA + 16;
#pragma unroll
        for (int j = 0; j < 8; ++j) {
            int row = q0 + j + 8 * half;
            float e0 = s0[j] * sc, e1 = s1[j] * sc;
            if (needMask) {
                if (colA > row) e0 = -__builtin_inff();
                if (colB > row) e1 = -__builtin_inff();
            }
            float rm = fmaxf(e0, e1);
#pragma unroll
            for (int msk = 1; msk < 16; msk <<= 1)
                rm = fmaxf(rm, __shfl_xor(rm, msk, 32));
            float mn = fmaxf(mi[j], rm);
            alph[j] = __expf(mi[j] - mn);
            mi[j] = mn;
            float p0 = __expf(e0 - mn), p1 = __expf(e1 - mn);
            float rs = p0 + p1;
#pragma unroll
            for (int msk = 1; msk < 16; msk <<= 1)
                rs += __shfl_xor(rs, msk, 32);
            li[j] = li[j] * alph[j] + rs;
            // P (16x32) into LDS, row-major for A-fragment reload
            pw[(j + 8 * half) * 32 + (lane & 15)]      = (f16)p0;
            pw[(j + 8 * half) * 32 + 16 + (lane & 15)] = (f16)p1;
        }
#pragma unroll
        for (int dt = 0; dt < 8; ++dt)
#pragma unroll
            for (int j = 0; j < 8; ++j) O[dt][j] *= alph[j];

        asm volatile("s_wait_dscnt 0x0" ::: "memory");
        v16h pa = load_frag_a(pw, 32);
#pragma unroll
        for (int dt = 0; dt < 8; ++dt) {
            v16h bv = load_frag_b(vv + (size_t)j0 * HD + dt * 16, HD);
            O[dt] = wmma_f16(pa, bv, O[dt]);
        }
    }

    float inv[8];
#pragma unroll
    for (int j = 0; j < 8; ++j) inv[j] = 1.0f / li[j];
#pragma unroll
    for (int dt = 0; dt < 8; ++dt) {
        int d = dt * 16 + (lane & 15);
#pragma unroll
        for (int j = 0; j < 8; ++j) {
            int row = q0 + j + 8 * half;
            float o = O[dt][j] * inv[j];
            a_h[((size_t)(b * SEQ + row)) * DIMN + h * HD + d] = (f16)o;
        }
    }
}

// ---------------------------------------------------------------------------
// Phase 3: output projection  out = attn @ wo^T   (f32 result to d_out)
// grid = (M/64, N/256); wave tile 32x64 as in phase 1.
// ---------------------------------------------------------------------------
__global__ __launch_bounds__(256) void oproj_kernel(const f16* __restrict__ ah,
                                                    const f16* __restrict__ woT,
                                                    float* __restrict__ out) {
    int lane = threadIdx.x & 31;
    int wave = threadIdx.x >> 5;
    int m0 = blockIdx.x * 64 + (wave & 1) * 32;
    int n0 = blockIdx.y * 256 + (wave >> 1) * 64;

    v8f acc[2][4] = {};
#pragma unroll 2
    for (int k0 = 0; k0 < DIMN; k0 += 32) {
        __builtin_prefetch(ah + (size_t)m0 * DIMN + k0 + 64, 0, 0);
        v16h a0 = load_frag_a(ah + (size_t)m0 * DIMN + k0, DIMN);
        v16h a1 = load_frag_a(ah + (size_t)(m0 + 16) * DIMN + k0, DIMN);
#pragma unroll
        for (int tj = 0; tj < 4; ++tj) {
            v16h bf = load_frag_b(woT + (size_t)k0 * DIMN + n0 + tj * 16, DIMN);
            acc[0][tj] = wmma_f16(a0, bf, acc[0][tj]);
            acc[1][tj] = wmma_f16(a1, bf, acc[1][tj]);
        }
    }
    int half = lane >> 4;
#pragma unroll
    for (int ti = 0; ti < 2; ++ti)
#pragma unroll
        for (int tj = 0; tj < 4; ++tj) {
            int col = n0 + tj * 16 + (lane & 15);
#pragma unroll
            for (int j = 0; j < 8; ++j) {
                int row = m0 + ti * 16 + j + 8 * half;
                out[(size_t)row * DIMN + col] = acc[ti][tj][j];
            }
        }
}

// ---------------------------------------------------------------------------
// Host-side launcher. Scratch layout (d_ws):
//   [0,16M)   xh      f16 4096x2048
//   [16,24M)  wqT     f16 2048x2048 (K x N)
//   [24,32M)  wkT
//   [32,40M)  wvT
//   [40,48M)  woT
//   [48,64M)  q_h     f16 [b][h][s][d]
//   [64,80M)  kT_h    f16 [b][h][d][s]
//   [80,96M)  v_h     f16 [b][h][s][d]
//   [96,112M) a_h     f16 [b*s][dim]
// ---------------------------------------------------------------------------
extern "C" void kernel_launch(void* const* d_in, const int* in_sizes, int n_in,
                              void* d_out, int out_size, void* d_ws, size_t ws_size,
                              hipStream_t stream) {
    (void)in_sizes; (void)n_in; (void)out_size; (void)ws_size;
    const float* x  = (const float*)d_in[0];
    // d_in[1] = mask (implied by causal masking in-kernel)
    const float* wq = (const float*)d_in[2];
    const float* wk = (const float*)d_in[3];
    const float* wv = (const float*)d_in[4];
    const float* wo = (const float*)d_in[5];
    float* out = (float*)d_out;

    char* ws = (char*)d_ws;
    const size_t MB = 1ull << 20;
    f16* xh   = (f16*)(ws + 0 * MB);
    f16* wqT  = (f16*)(ws + 16 * MB);
    f16* wkT  = (f16*)(ws + 24 * MB);
    f16* wvT  = (f16*)(ws + 32 * MB);
    f16* woT  = (f16*)(ws + 40 * MB);
    f16* q_h  = (f16*)(ws + 48 * MB);
    f16* kT_h = (f16*)(ws + 64 * MB);
    f16* v_h  = (f16*)(ws + 80 * MB);
    f16* a_h  = (f16*)(ws + 96 * MB);

    cvt_x_kernel<<<(MTOT * DIMN) / 256, 256, 0, stream>>>(x, xh);
    cvt_w_kernel<<<dim3(DIMN / 32, DIMN / 32, 4), dim3(32, 8), 0, stream>>>(
        wq, wk, wv, wo, wqT, wkT, wvT, woT);
    qkv_rope_kernel<<<dim3(MTOT / 64, DIMN / 256, 3), 256, 0, stream>>>(
        xh, wqT, wkT, wvT, q_h, kT_h, v_h);
    attn_kernel<<<dim3(SEQ / 128, HEADS, BATCH), 256, 0, stream>>>(
        q_h, kT_h, v_h, a_h);
    oproj_kernel<<<dim3(MTOT / 64, DIMN / 256), 256, 0, stream>>>(
        a_h, woT, out);
}
